// CausalAttention_49143015801084
// MI455X (gfx1250) — matrix operand order repaired and checked by reading, not compiled, not run on hardware
//
#include <hip/hip_runtime.h>

// ---------------------------------------------------------------------------
// Causal multi-head attention for MI455X (gfx1250, wave32, WMMA).
// fp32 inputs -> bf16 operands (pre-converted) -> v_wmma_f32_16x16x32_bf16.
// Attention: S^T = K*Q^T trick (in-lane softmax, register-resident P tile),
// double-buffered async global->LDS staging of K/V.
// ---------------------------------------------------------------------------

typedef __bf16 bf16_t;
typedef __attribute__((ext_vector_type(16))) __bf16 v16bf;
typedef __attribute__((ext_vector_type(8)))  __bf16 bf16x8;
typedef __attribute__((ext_vector_type(8)))  float  v8f;

#define B_    4
#define S_    2048
#define DIM_  1024
#define H_    16
#define HD_   64
#define SCALE_ 0.125f   // HD^-0.5

#if defined(__has_builtin)
#if __has_builtin(__builtin_amdgcn_global_load_async_to_lds_b128)
#define ASYNC_LDS 1
#endif
#endif

#ifdef ASYNC_LDS
// Probe-learned signature: param0 = int4 addrspace(1)*, param1 = addrspace(3)*,
// then imm offset, imm cpol.
typedef int v4i_raw __attribute__((vector_size(16)));
static __device__ inline void async_copy_b128(void* lds, const void* gsrc) {
  __builtin_amdgcn_global_load_async_to_lds_b128(
      (__attribute__((address_space(1))) v4i_raw*)(v4i_raw*)(void*)gsrc,
      (__attribute__((address_space(3))) v4i_raw*)(v4i_raw*)lds, 0, 0);
}
static __device__ inline void wait_async4() {   // allow 4 newer ops in flight
#if __has_builtin(__builtin_amdgcn_s_wait_asynccnt)
  __builtin_amdgcn_s_wait_asynccnt(4);
#else
  asm volatile("s_wait_asynccnt 4" ::: "memory");
#endif
}
static __device__ inline void wait_async0() {
#if __has_builtin(__builtin_amdgcn_s_wait_asynccnt)
  __builtin_amdgcn_s_wait_asynccnt(0);
#else
  asm volatile("s_wait_asynccnt 0" ::: "memory");
#endif
}
#endif

// ---- WMMA wrapper ---------------------------------------------------------
static __device__ inline v8f wmma_bf16(v16bf a, v16bf b, v8f c) {
  return __builtin_amdgcn_wmma_f32_16x16x32_bf16(false, a, false, b,
                                                 (short)0, c, false, false);
}

// ---- A fragment (16x32, MxK): lane<16 row m holds K {0..7,16..23},
//      lane>=16 same row, K {8..15,24..31}. Source row-major over K. --------
static __device__ inline v16bf fragA(const bf16_t* p, int ld, int lane) {
  const int r  = lane & 15;
  const int kh = (lane >> 4) << 3;            // 0 or 8
  const bf16_t* row = p + (size_t)r * ld + kh;
  bf16x8 lo = *reinterpret_cast<const bf16x8*>(row);
  bf16x8 hi = *reinterpret_cast<const bf16x8*>(row + 16);
  v16bf v;
#pragma unroll
  for (int i = 0; i < 8; ++i) { v[i] = lo[i]; v[i + 8] = hi[i]; }
  return v;
}

static __device__ inline v16bf fragA(const float* p, int ld, int lane) {
  const int r  = lane & 15;
  const int kh = (lane >> 4) << 3;
  const float* row = p + (size_t)r * ld + kh;
  v16bf v;
#pragma unroll
  for (int i = 0; i < 8; ++i) {
    v[i]     = (bf16_t)row[i];
    v[i + 8] = (bf16_t)row[16 + i];
  }
  return v;
}

// ---- B fragment (32x16, KxN): lane n holds K {0..7,16..23}, lane n+16 K {8..15,24..31}.
//      Source: "N-major" rows contiguous over K. ----------------------------
static __device__ inline v16bf fragB(const bf16_t* p, int ld, int lane) {
  const int r   = lane & 15;
  const int off = (lane >> 4) << 3;           // 0 or 8
  const bf16_t* row = p + (size_t)r * ld + off;
  bf16x8 lo = *reinterpret_cast<const bf16x8*>(row);
  bf16x8 hi = *reinterpret_cast<const bf16x8*>(row + 16);
  v16bf v;
#pragma unroll
  for (int i = 0; i < 8; ++i) { v[i] = lo[i]; v[i + 8] = hi[i]; }
  return v;
}

static __device__ inline v16bf fragB(const float* p, int ld, int lane) {
  const int r   = lane & 15;
  const int off = (lane >> 4) << 3;
  const float* row = p + (size_t)r * ld + off;
  v16bf v;
#pragma unroll
  for (int i = 0; i < 8; ++i) {
    v[i]     = (bf16_t)row[i];
    v[i + 8] = (bf16_t)row[16 + i];
  }
  return v;
}

// ---------------------------------------------------------------------------
// Elementwise fp32 -> bf16 (8 elems/thread, b128 loads).
// ---------------------------------------------------------------------------
__global__ __launch_bounds__(256) void f32_to_bf16_kernel(
    const float* __restrict__ x, bf16_t* __restrict__ y, int n) {
  const int i = (blockIdx.x * 256 + threadIdx.x) * 8;
  if (i + 7 < n) {
    const float4 a = *reinterpret_cast<const float4*>(x + i);
    const float4 b = *reinterpret_cast<const float4*>(x + i + 4);
    bf16x8 o;
    o[0] = (bf16_t)a.x; o[1] = (bf16_t)a.y; o[2] = (bf16_t)a.z; o[3] = (bf16_t)a.w;
    o[4] = (bf16_t)b.x; o[5] = (bf16_t)b.y; o[6] = (bf16_t)b.z; o[7] = (bf16_t)b.w;
    *reinterpret_cast<bf16x8*>(y + i) = o;
  }
}

// ---------------------------------------------------------------------------
// Projection GEMM: Y[M,N] = X[M,K] @ W[N,K]^T -> bf16 head-split output.
// Block = 256 thr = 8 waves (4Mx2N), wave tile 32x64, block tile 128x128.
// ---------------------------------------------------------------------------
template <typename TA, typename TB>
__global__ __launch_bounds__(256) void proj_gemm_kernel(
    const TA* __restrict__ X, const TB* __restrict__ W,
    bf16_t* __restrict__ Y, int vTranspose) {
  const int lane  = threadIdx.x & 31;
  const int wave  = threadIdx.x >> 5;
  const int mbase = blockIdx.x * 128 + (wave & 3) * 32;
  const int nbase = blockIdx.y * 128 + (wave >> 2) * 64;

  v8f acc[2][4] = {{{}, {}, {}, {}}, {{}, {}, {}, {}}};
  for (int k0 = 0; k0 < DIM_; k0 += 32) {
    const v16bf a0 = fragA(X + (size_t)mbase * DIM_ + k0,        DIM_, lane);
    const v16bf a1 = fragA(X + (size_t)(mbase + 16) * DIM_ + k0, DIM_, lane);
#pragma unroll
    for (int f = 0; f < 4; ++f) {
      const v16bf b = fragB(W + (size_t)(nbase + f * 16) * DIM_ + k0, DIM_, lane);
      acc[0][f] = wmma_bf16(a0, b, acc[0][f]);
      acc[1][f] = wmma_bf16(a1, b, acc[1][f]);
    }
  }

  const int n0 = lane & 15;
  const int mo = (lane >> 4) << 3;            // C layout: VGPR v -> row v / v+8
#pragma unroll
  for (int i = 0; i < 2; ++i) {
#pragma unroll
    for (int f = 0; f < 4; ++f) {
#pragma unroll
      for (int v = 0; v < 8; ++v) {
        const int m  = mbase + i * 16 + mo + v;
        const int n  = nbase + f * 16 + n0;
        const int bb = m >> 11, s  = m & (S_ - 1);
        const int h  = n >> 6,  hd = n & (HD_ - 1);
        const size_t idx = vTranspose
            ? ((size_t)((bb * H_ + h) * HD_ + hd) * S_ + s)
            : ((size_t)((bb * H_ + h) * S_  + s ) * HD_ + hd);
        Y[idx] = (bf16_t)acc[i][f][v];
      }
    }
  }
}

// ---------------------------------------------------------------------------
// Flash-style causal attention, S^T formulation.
// Block = 4 waves = 64 query rows of one (b,h). Double-buffered K/V staging
// in LDS via async global->LDS. Per wave: S^T = K*Q^T (lane = query), so
// softmax stats are in-lane reductions + one xor-16 shuffle, and the P tile
// lands directly in the A-operand register layout for P@V.
// ---------------------------------------------------------------------------
__global__ __launch_bounds__(128) void attn_fa_kernel(
    const bf16_t* __restrict__ Qb, const bf16_t* __restrict__ Kb,
    const bf16_t* __restrict__ Vt, bf16_t* __restrict__ Cb) {
  __shared__ bf16_t Kt[2][32][64];            // 2 x 4 KB: keys x d
  __shared__ bf16_t Vs[2][64][32];            // 2 x 4 KB: d x keys

  const int lane   = threadIdx.x & 31;
  const int wave   = threadIdx.x >> 5;
  const int qblk   = blockIdx.x & (S_ / 64 - 1);
  const int bh     = blockIdx.x >> 5;         // 32 q-blocks per (b,h)
  const int qbase0 = qblk * 64;
  const int qbase  = qbase0 + wave * 16;

  const bf16_t* Qp = Qb + ((size_t)bh * S_ + qbase) * HD_;
  const bf16_t* Kp = Kb + (size_t)bh * S_ * HD_;
  const bf16_t* Vp = Vt + (size_t)bh * HD_ * S_;

  // Q as B-operand fragments (lane = query column of S^T).
  const v16bf qb0 = fragB(Qp,      HD_, lane);   // d = 0..31
  const v16bf qb1 = fragB(Qp + 32, HD_, lane);   // d = 32..63

  v8f acc0 = {}, acc1 = {}, acc2 = {}, acc3 = {};
  float m = -1.0e30f, l = 0.0f;               // stats for query (lane&15)

  const int n0  = lane & 15;
  const int mo  = (lane >> 4) << 3;           // acc row offset (O layout)
  const int qg  = qbase + n0;                 // this lane's query index
  const int kof = (lane >> 4) << 3;           // S^T key offset for this half

  const int kendWave = qbase + 16;
  const int kendBlk  = qbase0 + 64;

  auto stage = [&](int kb, int ibuf) {
    const int t = threadIdx.x;
    const bf16_t* gk = Kp + (size_t)(kb + (t >> 2)) * HD_ + (t & 3) * 16;
    bf16_t*       lk = &Kt[ibuf][t >> 2][(t & 3) * 16];
    const bf16_t* gv = Vp + (size_t)(t >> 1) * S_ + kb + (t & 1) * 16;
    bf16_t*       lv = &Vs[ibuf][t >> 1][(t & 1) * 16];
#ifdef ASYNC_LDS
    async_copy_b128(lk,     gk);
    async_copy_b128(lk + 8, gk + 8);
    async_copy_b128(lv,     gv);
    async_copy_b128(lv + 8, gv + 8);
#else
    *reinterpret_cast<bf16x8*>(lk)     = *reinterpret_cast<const bf16x8*>(gk);
    *reinterpret_cast<bf16x8*>(lk + 8) = *reinterpret_cast<const bf16x8*>(gk + 8);
    *reinterpret_cast<bf16x8*>(lv)     = *reinterpret_cast<const bf16x8*>(gv);
    *reinterpret_cast<bf16x8*>(lv + 8) = *reinterpret_cast<const bf16x8*>(gv + 8);
#endif
  };

  stage(0, 0);
  int ibuf = 0;
  for (int kb = 0; kb < kendBlk; kb += 32, ibuf ^= 1) {
    const bool more = (kb + 32) < kendBlk;
    if (more) stage(kb + 32, ibuf ^ 1);       // prefetch next block
#ifdef ASYNC_LDS
    if (more) wait_async4(); else wait_async0();  // current block landed
#endif
    __syncthreads();

    if (kb < kendWave) {
      // ---- S^T = K * Q^T : rows = keys, cols = queries (2 x 16-key tiles).
      const v16bf ka00 = fragA(&Kt[ibuf][0][0],   64, lane);
      const v16bf ka01 = fragA(&Kt[ibuf][0][32],  64, lane);
      const v16bf ka10 = fragA(&Kt[ibuf][16][0],  64, lane);
      const v16bf ka11 = fragA(&Kt[ibuf][16][32], 64, lane);
      v8f st0 = {}, st1 = {};
      st0 = wmma_bf16(ka00, qb0, st0);
      st0 = wmma_bf16(ka01, qb1, st0);
      st1 = wmma_bf16(ka10, qb0, st1);
      st1 = wmma_bf16(ka11, qb1, st1);

      // ---- Scale + causal mask + in-lane row max.
      float s0v[8], s1v[8];
      float mx = -1.0e30f;
#pragma unroll
      for (int v = 0; v < 8; ++v) {
        const int kg0 = kb + kof + v;         // tile0 key for this element
        float a = st0[v] * SCALE_;
        float b = st1[v] * SCALE_;
        if (kg0      > qg) a = -1.0e9f;
        if (kg0 + 16 > qg) b = -1.0e9f;
        s0v[v] = a; s1v[v] = b;
        mx = fmaxf(mx, fmaxf(a, b));
      }
      mx = fmaxf(mx, __shfl_xor(mx, 16, 32)); // merge the two lane halves
      const float mnew = fmaxf(m, mx);
      const float corr = __expf(m - mnew);
      m = mnew;

      // ---- exp + in-lane row sum; P packs directly into the A fragment:
      // lane<16 holds keys {0..7,16..23}, lane>=16 holds {8..15,24..31}.
      v16bf pa;
      float rs = 0.0f;
#pragma unroll
      for (int v = 0; v < 8; ++v) {
        const float e0 = __expf(s0v[v] - mnew);
        const float e1 = __expf(s1v[v] - mnew);
        rs += e0 + e1;
        pa[v]     = (bf16_t)e0;
        pa[v + 8] = (bf16_t)e1;
      }
      rs += __shfl_xor(rs, 16, 32);
      l = l * corr + rs;

      // ---- Rescale O accumulator: row mo+v needs corr from lane mo+v.
#pragma unroll
      for (int v = 0; v < 8; ++v) {
        const float c = __shfl(corr, mo + v, 32);
        acc0[v] *= c; acc1[v] *= c; acc2[v] *= c; acc3[v] *= c;
      }

      // ---- O += P @ V (V^T rows in LDS are the B-operand columns).
      acc0 = wmma_bf16(pa, fragB(&Vs[ibuf][0][0],  32, lane), acc0);
      acc1 = wmma_bf16(pa, fragB(&Vs[ibuf][16][0], 32, lane), acc1);
      acc2 = wmma_bf16(pa, fragB(&Vs[ibuf][32][0], 32, lane), acc2);
      acc3 = wmma_bf16(pa, fragB(&Vs[ibuf][48][0], 32, lane), acc3);
    }
    __syncthreads();   // everyone done reading this buffer
  }

  // ---- Normalize: 1/l for row mo+v comes from lane mo+v.
  const float linv = 1.0f / fmaxf(l, 1e-9f);
  float inv[8];
#pragma unroll
  for (int v = 0; v < 8; ++v) inv[v] = __shfl(linv, mo + v, 32);

  const int bb = bh >> 4;
  const int h  = bh & 15;
#pragma unroll
  for (int v = 0; v < 8; ++v) {
    const size_t base =
        (size_t)(bb * S_ + qbase + mo + v) * DIM_ + h * HD_ + n0;
    Cb[base +  0] = (bf16_t)(acc0[v] * inv[v]);
    Cb[base + 16] = (bf16_t)(acc1[v] * inv[v]);
    Cb[base + 32] = (bf16_t)(acc2[v] * inv[v]);
    Cb[base + 48] = (bf16_t)(acc3[v] * inv[v]);
  }
}

// ---------------------------------------------------------------------------
// Output projection: out[M,N] = Cb[M,K](bf16) @ Wo[N,K]^T -> fp32.
// ---------------------------------------------------------------------------
template <typename TB>
__global__ __launch_bounds__(256) void out_proj_kernel(
    const bf16_t* __restrict__ Cb, const TB* __restrict__ Wo,
    float* __restrict__ out) {
  const int lane  = threadIdx.x & 31;
  const int wave  = threadIdx.x >> 5;
  const int mbase = blockIdx.x * 128 + (wave & 3) * 32;
  const int nbase = blockIdx.y * 128 + (wave >> 2) * 64;

  v8f acc[2][4] = {{{}, {}, {}, {}}, {{}, {}, {}, {}}};
  for (int k0 = 0; k0 < DIM_; k0 += 32) {
    const v16bf a0 = fragA(Cb + (size_t)mbase * DIM_ + k0,        DIM_, lane);
    const v16bf a1 = fragA(Cb + (size_t)(mbase + 16) * DIM_ + k0, DIM_, lane);
#pragma unroll
    for (int f = 0; f < 4; ++f) {
      const v16bf b = fragB(Wo + (size_t)(nbase + f * 16) * DIM_ + k0, DIM_, lane);
      acc[0][f] = wmma_bf16(a0, b, acc[0][f]);
      acc[1][f] = wmma_bf16(a1, b, acc[1][f]);
    }
  }

  const int n0 = lane & 15;
  const int mo = (lane >> 4) << 3;
#pragma unroll
  for (int i = 0; i < 2; ++i) {
#pragma unroll
    for (int f = 0; f < 4; ++f) {
#pragma unroll
      for (int v = 0; v < 8; ++v) {
        const int m = mbase + i * 16 + mo + v;
        const int n = nbase + f * 16 + n0;
        out[(size_t)m * DIM_ + n] = acc[i][f][v];
      }
    }
  }
}

// ---------------------------------------------------------------------------
// Launch. Workspace layout (bf16 elements):
//   Qb | Kb | Vt | Cb                 (always, 4 * 8.39M = 67.1 MB)
//   Xq | Xk | Xv | Wq | Wk | Wv | Wo  (fast path only, +58.7 MB)
// ---------------------------------------------------------------------------
extern "C" void kernel_launch(void* const* d_in, const int* in_sizes, int n_in,
                              void* d_out, int out_size, void* d_ws, size_t ws_size,
                              hipStream_t stream) {
  (void)in_sizes; (void)n_in; (void)out_size;
  const float* query = (const float*)d_in[0];
  const float* key   = (const float*)d_in[1];
  const float* value = (const float*)d_in[2];
  // d_in[3]: causal (tril) mask -- implicit in the attention kernel.
  const float* Wq = (const float*)d_in[4];
  const float* Wk = (const float*)d_in[5];
  const float* Wv = (const float*)d_in[6];
  const float* Wo = (const float*)d_in[7];

  const size_t elems = (size_t)B_ * S_ * DIM_;     // 8,388,608
  const size_t welems = (size_t)DIM_ * DIM_;       // 1,048,576
  bf16_t* Qb = (bf16_t*)d_ws;
  bf16_t* Kb = Qb + elems;
  bf16_t* Vt = Kb + elems;
  bf16_t* Cb = Vt + elems;

  const size_t need_fast = (4 * elems + 3 * elems + 4 * welems) * sizeof(bf16_t);
  const dim3 gg(B_ * S_ / 128, DIM_ / 128);        // 64 x 8 blocks, 256 thr
  const int nAttnBlk = B_ * H_ * (S_ / 64);        // 2048 blocks, 128 thr

  if (ws_size >= need_fast) {
    // ---- Fast path: pre-convert activations + weights to bf16 once.
    bf16_t* Xq  = Cb + elems;
    bf16_t* Xk  = Xq + elems;
    bf16_t* Xv  = Xk + elems;
    bf16_t* Wqb = Xv + elems;
    bf16_t* Wkb = Wqb + welems;
    bf16_t* Wvb = Wkb + welems;
    bf16_t* Wob = Wvb + welems;

    const int gx = (int)(elems / 2048), gw = (int)(welems / 2048);
    f32_to_bf16_kernel<<<gx, 256, 0, stream>>>(query, Xq, (int)elems);
    f32_to_bf16_kernel<<<gx, 256, 0, stream>>>(key,   Xk, (int)elems);
    f32_to_bf16_kernel<<<gx, 256, 0, stream>>>(value, Xv, (int)elems);
    f32_to_bf16_kernel<<<gw, 256, 0, stream>>>(Wq, Wqb, (int)welems);
    f32_to_bf16_kernel<<<gw, 256, 0, stream>>>(Wk, Wkb, (int)welems);
    f32_to_bf16_kernel<<<gw, 256, 0, stream>>>(Wv, Wvb, (int)welems);
    f32_to_bf16_kernel<<<gw, 256, 0, stream>>>(Wo, Wob, (int)welems);

    proj_gemm_kernel<bf16_t, bf16_t><<<gg, 256, 0, stream>>>(Xq, Wqb, Qb, 0);
    proj_gemm_kernel<bf16_t, bf16_t><<<gg, 256, 0, stream>>>(Xk, Wkb, Kb, 0);
    proj_gemm_kernel<bf16_t, bf16_t><<<gg, 256, 0, stream>>>(Xv, Wvb, Vt, 1);
    attn_fa_kernel<<<nAttnBlk, 128, 0, stream>>>(Qb, Kb, Vt, Cb);
    out_proj_kernel<bf16_t><<<gg, 256, 0, stream>>>(Cb, Wob, (float*)d_out);
  } else {
    // ---- Fallback: convert fp32 operands in-register inside the GEMMs.
    proj_gemm_kernel<float, float><<<gg, 256, 0, stream>>>(query, Wq, Qb, 0);
    proj_gemm_kernel<float, float><<<gg, 256, 0, stream>>>(key,   Wk, Kb, 0);
    proj_gemm_kernel<float, float><<<gg, 256, 0, stream>>>(value, Wv, Vt, 1);
    attn_fa_kernel<<<nAttnBlk, 128, 0, stream>>>(Qb, Kb, Vt, Cb);
    out_proj_kernel<float><<<gg, 256, 0, stream>>>(Cb, Wo, (float*)d_out);
  }
}
